// Up_19791209299985
// MI455X (gfx1250) — matrix operand order repaired and checked by reading, not compiled, not run on hardware
//
#include <hip/hip_runtime.h>
#include <hip/hip_bf16.h>

typedef __attribute__((ext_vector_type(16))) _Float16 v16h;
typedef __attribute__((ext_vector_type(8)))  _Float16 v8h;
typedef __attribute__((ext_vector_type(8)))  float    v8f;

// ---------------------------------------------------------------------------
// Weight prepack: W f32 [N][K] row-major -> f16 packed WMMA-B layout.
// Tile (kb,nb) = 512 halfs at ((kb*(N/16)+nb)*512); within tile: lane*16+j,
// value = W[n][k] with n = nb*16 + (lane&15), k = kb*32 + 16*(lane>>4) + j.
// ---------------------------------------------------------------------------
__global__ void prepack_b_k(const float* __restrict__ W, _Float16* __restrict__ Bp,
                            int K, int N) {
  int idx = blockIdx.x * 256 + threadIdx.x;
  if (idx >= K * N) return;
  int j    = idx & 15;
  int lane = (idx >> 4) & 31;
  int t    = idx >> 9;
  int nTiles = N >> 4;
  int kb = t / nTiles, nb = t % nTiles;
  int k = kb * 32 + 16 * (lane >> 4) + j;
  int n = nb * 16 + (lane & 15);
  Bp[idx] = (_Float16)W[(size_t)n * K + k];
}

__global__ void pack_bias576_k(const float* __restrict__ qb, const float* __restrict__ vb,
                               float* __restrict__ out) {
  int i = threadIdx.x;  // 576 threads
  out[i] = (i < 192) ? qb[i] : ((i < 384) ? 0.0f : vb[i - 384]);
}

// ---------------------------------------------------------------------------
// Bilinear 2x upsample (align_corners), f32 (B,256,64,64) -> f16 [b][y][x][ci]
// ---------------------------------------------------------------------------
__global__ void upsample_k(const float* __restrict__ x1, _Float16* __restrict__ xu) {
  int idx = blockIdx.x * 256 + threadIdx.x;   // 4*128*128*256 total
  int ci = idx & 255;
  int x  = (idx >> 8) & 127;
  int y  = (idx >> 15) & 127;
  int bb = idx >> 22;
  float fy = y * (63.0f / 127.0f);
  int y0 = (int)fy; float wy = fy - (float)y0; int y1 = (y0 + 1 < 63) ? y0 + 1 : 63;
  float fx = x * (63.0f / 127.0f);
  int x0 = (int)fx; float wx = fx - (float)x0; int x1i = (x0 + 1 < 63) ? x0 + 1 : 63;
  const float* p = x1 + ((size_t)(bb * 256 + ci)) * 4096;
  float v00 = p[y0 * 64 + x0], v01 = p[y0 * 64 + x1i];
  float v10 = p[y1 * 64 + x0], v11 = p[y1 * 64 + x1i];
  float v = (v00 * (1.0f - wx) + v01 * wx) * (1.0f - wy)
          + (v10 * (1.0f - wx) + v11 * wx) * wy;
  xu[idx] = (_Float16)v;
}

// ---------------------------------------------------------------------------
// im2col for patch-embed: Ape[token][4096], k = c*16 + kh*4 + kw
// c<128 from x2 f32 (B,128,128,128); c>=128 from x1up f16 [b*16384+y*128+x][128]
// ---------------------------------------------------------------------------
__global__ void im2col_k(const float* __restrict__ x2, const _Float16* __restrict__ x1up,
                         _Float16* __restrict__ Ape) {
  int idx = blockIdx.x * 256 + threadIdx.x;  // 4096*4096 total
  int k   = idx & 4095;
  int tok = idx >> 12;
  int pw = tok & 31, ph = (tok >> 5) & 31, bb = tok >> 10;
  int kw = k & 3, kh = (k >> 2) & 3, c = k >> 4;
  int y = ph * 4 + kh, x = pw * 4 + kw;
  float v;
  if (c < 128)
    v = x2[(((size_t)(bb * 128 + c)) * 128 + y) * 128 + x];
  else
    v = (float)x1up[((size_t)(bb * 16384 + y * 128 + x)) * 128 + (c - 128)];
  Ape[idx] = (_Float16)v;
}

// ---------------------------------------------------------------------------
// WMMA GEMM with A-fragment reuse: one wave computes a 16 x (NT*16) strip.
// A[M][K] f16 row-major; Bp packed. Per K-step: issue A + all NT B-tile loads
// as one batch (single s_clause, partial waits), then run the WMMA chain so
// later loads overlap earlier WMMAs within the wave.
// outMode: 0 = f32 [M][N]; 1 = f16 [M][N]; 2 = f32 NCHW scatter (B,128,32,32)
// act: 0 none, 1 exact GELU
// ---------------------------------------------------------------------------
template <int NT>
__global__ void gemm_wmma_k(const _Float16* __restrict__ A, const _Float16* __restrict__ Bp,
                            const float* __restrict__ bias, void* __restrict__ Cout,
                            int M, int K, int N, int act, int outMode) {
  int mt = blockIdx.x;
  int ntBase = blockIdx.y * NT;
  int lane = threadIdx.x & 31;
  int hf = lane >> 4, lm = lane & 15;
  const _Float16* Arow = A + (size_t)(mt * 16 + lm) * K;
  int nTiles = N >> 4;
  const _Float16* Bl = Bp + (size_t)ntBase * 512 + (size_t)lane * 16;
  v8f acc[NT];
#pragma unroll
  for (int q = 0; q < NT; ++q) acc[q] = (v8f){};
  int kSteps = K >> 5;
  for (int kb = 0; kb < kSteps; ++kb) {
    // ---- batch-issue all loads for this K-step ----
    v8h a0 = *(const v8h*)(Arow + kb * 32 + 16 * hf);
    v8h a1 = *(const v8h*)(Arow + kb * 32 + 16 * hf + 8);
    const _Float16* bp = Bl + (size_t)kb * nTiles * 512;
    v8h b0a[NT], b1a[NT];
#pragma unroll
    for (int q = 0; q < NT; ++q) {
      b0a[q] = *(const v8h*)(bp + q * 512);
      b1a[q] = *(const v8h*)(bp + q * 512 + 8);
    }
    // ---- WMMA chain (loads for later q still in flight) ----
    v16h av;
#pragma unroll
    for (int j = 0; j < 8; ++j) { av[j] = a0[j]; av[8 + j] = a1[j]; }
#pragma unroll
    for (int q = 0; q < NT; ++q) {
      v16h bv;
#pragma unroll
      for (int j = 0; j < 8; ++j) { bv[j] = b0a[q][j]; bv[8 + j] = b1a[q][j]; }
      acc[q] = __builtin_amdgcn_wmma_f32_16x16x32_f16(false, av, false, bv, (short)0,
                                                      acc[q], false, false);
    }
  }
#pragma unroll
  for (int q = 0; q < NT; ++q) {
    int n = (ntBase + q) * 16 + lm;
    float bvl = bias ? bias[n] : 0.0f;
#pragma unroll
    for (int r = 0; r < 8; ++r) {
      int m = mt * 16 + r + 8 * hf;
      float v = acc[q][r] + bvl;
      if (act == 1) v = 0.5f * v * (1.0f + erff(v * 0.70710678118f));
      if (outMode == 0)      ((float*)Cout)[(size_t)m * N + n] = v;
      else if (outMode == 1) ((_Float16*)Cout)[(size_t)m * N + n] = (_Float16)v;
      else {
        int bb = m >> 10, t = m & 1023;
        ((float*)Cout)[(size_t)bb * 131072 + (size_t)n * 1024 + t] = v;
      }
    }
  }
}

// ---------------------------------------------------------------------------
// Residual + LayerNorm over 192 dims. residual: X = X + LN(P) else X = LN(P).
// Writes X (f32) and Xh (f16 GEMM-A copy). 64 threads, 3 dims each.
// ---------------------------------------------------------------------------
__global__ void addln_k(const float* __restrict__ P, float* __restrict__ X,
                        _Float16* __restrict__ Xh, const float* __restrict__ g,
                        const float* __restrict__ bt, int residual) {
  int t = blockIdx.x, tid = threadIdx.x;
  __shared__ float red[64];
  float v0 = P[(size_t)t * 192 + tid];
  float v1 = P[(size_t)t * 192 + 64 + tid];
  float v2 = P[(size_t)t * 192 + 128 + tid];
  red[tid] = v0 + v1 + v2; __syncthreads();
  for (int o = 32; o > 0; o >>= 1) { if (tid < o) red[tid] += red[tid + o]; __syncthreads(); }
  float mu = red[0] * (1.0f / 192.0f); __syncthreads();
  float d0 = v0 - mu, d1 = v1 - mu, d2 = v2 - mu;
  red[tid] = d0 * d0 + d1 * d1 + d2 * d2; __syncthreads();
  for (int o = 32; o > 0; o >>= 1) { if (tid < o) red[tid] += red[tid + o]; __syncthreads(); }
  float rs = rsqrtf(red[0] * (1.0f / 192.0f) + 1e-5f);
#pragma unroll
  for (int q = 0; q < 3; ++q) {
    int d = tid + 64 * q;
    float vv = (q == 0 ? v0 : (q == 1 ? v1 : v2));
    float xn = (vv - mu) * rs * g[d] + bt[d];
    float xv = residual ? (X[(size_t)t * 192 + d] + xn) : xn;
    X[(size_t)t * 192 + d] = xv;
    Xh[(size_t)t * 192 + d] = (_Float16)xv;
  }
}

// ---------------------------------------------------------------------------
// Continuous-position-bias MLP: 225 rel-coords -> 512 hidden (ReLU) -> 6 heads
// ---------------------------------------------------------------------------
__global__ void cpb_hidden_k(const float* __restrict__ w1, const float* __restrict__ b1,
                             const float* __restrict__ w2, float* __restrict__ out225) {
  int r = blockIdx.x;       // 0..224
  int j = threadIdx.x;      // 0..511
  int dh = r / 15, dw = r % 15;
  float a0 = (float)(dh - 7) * (8.0f / 7.0f);
  float a1 = (float)(dw - 7) * (8.0f / 7.0f);
  auto warp_ = [](float a) {
    float s = (a > 0.0f) ? 1.0f : ((a < 0.0f) ? -1.0f : 0.0f);
    return s * log2f(fabsf(a) + 1.0f) * (1.0f / 3.0f);
  };
  float t0 = warp_(a0), t1 = warp_(a1);
  float h = fmaxf(t0 * w1[j * 2] + t1 * w1[j * 2 + 1] + b1[j], 0.0f);
  __shared__ float hid[512];
  hid[j] = h; __syncthreads();
  if (j < 6) {
    float s = 0.0f;
    for (int u = 0; u < 512; ++u) s += hid[u] * w2[j * 512 + u];
    out225[r * 6 + j] = 16.0f / (1.0f + expf(-s));
  }
}

__global__ void rpb_expand_k(const float* __restrict__ out225, float* __restrict__ rpb) {
  int idx = blockIdx.x * 256 + threadIdx.x;  // 6*64*64
  int j = idx & 63, i = (idx >> 6) & 63, h = idx >> 12;
  int dh = (i >> 3) - (j >> 3) + 7;
  int dw = (i & 7) - (j & 7) + 7;
  rpb[idx] = out225[(dh * 15 + dw) * 6 + h];
}

// ---------------------------------------------------------------------------
// Windowed cosine attention, one (window, head) per 64-thread block (2 waves).
// qkv f32 [4096][576] -> attnH f16 [4096][192] (handles shift roll/unroll).
// rpb + shift-mask are applied in the softmax pass (row-contiguous reads),
// keeping the WMMA loops free of global loads / branches.
// ---------------------------------------------------------------------------
template <int SHIFT>
__global__ void attn_k(const float* __restrict__ qkv, const float* __restrict__ rpb,
                       const float* __restrict__ logit_scale, _Float16* __restrict__ attnH) {
  __shared__ _Float16 qs[64 * 32];
  __shared__ _Float16 ks[64 * 32];
  __shared__ _Float16 vts[32 * 64];
  __shared__ float    Sm[64 * 64];
  __shared__ _Float16 ps[64 * 64];

  int blk = blockIdx.x;
  int h   = blk % 6;
  int win = blk / 6;
  int ww = win & 3, wh = (win >> 2) & 3, bb = win >> 4;
  int t = threadIdx.x;  // 0..63

  // ---- load q,k,v for token t of this window ----
  {
    int ih = t >> 3, iw = t & 7;
    int oh = (wh * 8 + ih + SHIFT) & 31;
    int ow = (ww * 8 + iw + SHIFT) & 31;
    int tok = bb * 1024 + oh * 32 + ow;
    const float* base = qkv + (size_t)tok * 576 + h * 32;
    float tmp[32]; float nrm; float inv;
    nrm = 0.0f;
#pragma unroll
    for (int d = 0; d < 32; ++d) { tmp[d] = base[d]; nrm += tmp[d] * tmp[d]; }
    inv = 1.0f / fmaxf(sqrtf(nrm), 1e-12f);
#pragma unroll
    for (int d = 0; d < 32; ++d) qs[t * 32 + d] = (_Float16)(tmp[d] * inv);
    nrm = 0.0f;
#pragma unroll
    for (int d = 0; d < 32; ++d) { tmp[d] = base[192 + d]; nrm += tmp[d] * tmp[d]; }
    inv = 1.0f / fmaxf(sqrtf(nrm), 1e-12f);
#pragma unroll
    for (int d = 0; d < 32; ++d) ks[t * 32 + d] = (_Float16)(tmp[d] * inv);
#pragma unroll
    for (int d = 0; d < 32; ++d) vts[d * 64 + t] = (_Float16)base[384 + d];
  }
  __syncthreads();

  float scale = expf(fminf(logit_scale[h], 4.6051702f));
  int wave = t >> 5, lane = t & 31, hf = lane >> 4, lm = lane & 15;

  // ---- S = scale * q^ k^T, 16x16 WMMA tiles (raw scores to LDS) ----
  for (int tile = 0; tile < 16; ++tile) {
    if ((tile & 1) != wave) continue;  // wave-uniform
    int mt = tile >> 2, nt = tile & 3;
    v8h a0 = *(const v8h*)(qs + (mt * 16 + lm) * 32 + 16 * hf);
    v8h a1 = *(const v8h*)(qs + (mt * 16 + lm) * 32 + 16 * hf + 8);
    v8h b0 = *(const v8h*)(ks + (nt * 16 + lm) * 32 + 16 * hf);
    v8h b1 = *(const v8h*)(ks + (nt * 16 + lm) * 32 + 16 * hf + 8);
    v16h av, bv;
#pragma unroll
    for (int j = 0; j < 8; ++j) { av[j] = a0[j]; av[8 + j] = a1[j]; bv[j] = b0[j]; bv[8 + j] = b1[j]; }
    v8f c = {};
    c = __builtin_amdgcn_wmma_f32_16x16x32_f16(false, av, false, bv, (short)0, c, false, false);
#pragma unroll
    for (int r = 0; r < 8; ++r) {
      int i = mt * 16 + r + 8 * hf;
      int j = nt * 16 + lm;
      Sm[i * 64 + j] = c[r] * scale;
    }
  }
  __syncthreads();

  // ---- softmax row t, fusing rpb (+mask): coalesced row-major rpb reads ----
  {
    const float* rrow = rpb + ((size_t)h * 64 + t) * 64;
    auto rg = [](int cc) { return cc < 24 ? 0 : (cc < 28 ? 1 : 2); };
    int idi = 0;
    if (SHIFT > 0) idi = rg(wh * 8 + (t >> 3)) * 3 + rg(ww * 8 + (t & 7));
    float mx = -1e30f;
    for (int j = 0; j < 64; ++j) {
      float s = Sm[t * 64 + j] + rrow[j];
      if (SHIFT > 0) {
        int idj = rg(wh * 8 + (j >> 3)) * 3 + rg(ww * 8 + (j & 7));
        s += (idi == idj) ? 0.0f : -100.0f;
      }
      Sm[t * 64 + j] = s;
      mx = fmaxf(mx, s);
    }
    float sum = 0.0f;
    for (int j = 0; j < 64; ++j) { float e = expf(Sm[t * 64 + j] - mx); sum += e; Sm[t * 64 + j] = e; }
    float rr = 1.0f / sum;
    for (int j = 0; j < 64; ++j) ps[t * 64 + j] = (_Float16)(Sm[t * 64 + j] * rr);
  }
  __syncthreads();

  // ---- O = P @ V (M=64, K=64, N=32) ----
  for (int tile = 0; tile < 8; ++tile) {
    if ((tile & 1) != wave) continue;
    int mt = tile >> 1, nt = tile & 1;
    v8f c = {};
#pragma unroll
    for (int kb = 0; kb < 2; ++kb) {
      v8h a0 = *(const v8h*)(ps + (mt * 16 + lm) * 64 + kb * 32 + 16 * hf);
      v8h a1 = *(const v8h*)(ps + (mt * 16 + lm) * 64 + kb * 32 + 16 * hf + 8);
      v8h b0 = *(const v8h*)(vts + (nt * 16 + lm) * 64 + kb * 32 + 16 * hf);
      v8h b1 = *(const v8h*)(vts + (nt * 16 + lm) * 64 + kb * 32 + 16 * hf + 8);
      v16h av, bv;
#pragma unroll
      for (int j = 0; j < 8; ++j) { av[j] = a0[j]; av[8 + j] = a1[j]; bv[j] = b0[j]; bv[8 + j] = b1[j]; }
      c = __builtin_amdgcn_wmma_f32_16x16x32_f16(false, av, false, bv, (short)0, c, false, false);
    }
#pragma unroll
    for (int r = 0; r < 8; ++r) {
      int i = mt * 16 + r + 8 * hf;
      int d = nt * 16 + lm;
      int oh = (wh * 8 + (i >> 3) + SHIFT) & 31;
      int ow = (ww * 8 + (i & 7) + SHIFT) & 31;
      int tok = bb * 1024 + oh * 32 + ow;
      attnH[(size_t)tok * 192 + h * 32 + d] = (_Float16)c[r];
    }
  }
}

// ===========================================================================
extern "C" void kernel_launch(void* const* d_in, const int* in_sizes, int n_in,
                              void* d_out, int out_size, void* d_ws, size_t ws_size,
                              hipStream_t stream) {
  (void)in_sizes; (void)n_in; (void)out_size; (void)ws_size;
  int ii = 0;
  const float* x1   = (const float*)d_in[ii++];
  const float* x2   = (const float*)d_in[ii++];
  const float* up_w = (const float*)d_in[ii++];
  const float* up_b = (const float*)d_in[ii++];
  const float* pe_w = (const float*)d_in[ii++];
  const float* pe_b = (const float*)d_in[ii++];
  const float* pe_g = (const float*)d_in[ii++];
  const float* pe_bn= (const float*)d_in[ii++];
  struct Blk {
    const float *qkv_w, *q_bias, *v_bias, *logit_scale, *cpb_w1, *cpb_b1, *cpb_w2,
                *proj_w, *proj_b, *n1g, *n1b, *fc1_w, *fc1_b, *fc2_w, *fc2_b, *n2g, *n2b;
  } blk[2];
  for (int b = 0; b < 2; ++b) {
    blk[b].qkv_w = (const float*)d_in[ii++]; blk[b].q_bias = (const float*)d_in[ii++];
    blk[b].v_bias = (const float*)d_in[ii++]; blk[b].logit_scale = (const float*)d_in[ii++];
    blk[b].cpb_w1 = (const float*)d_in[ii++]; blk[b].cpb_b1 = (const float*)d_in[ii++];
    blk[b].cpb_w2 = (const float*)d_in[ii++]; blk[b].proj_w = (const float*)d_in[ii++];
    blk[b].proj_b = (const float*)d_in[ii++]; blk[b].n1g = (const float*)d_in[ii++];
    blk[b].n1b = (const float*)d_in[ii++];   blk[b].fc1_w = (const float*)d_in[ii++];
    blk[b].fc1_b = (const float*)d_in[ii++]; blk[b].fc2_w = (const float*)d_in[ii++];
    blk[b].fc2_b = (const float*)d_in[ii++]; blk[b].n2g = (const float*)d_in[ii++];
    blk[b].n2b = (const float*)d_in[ii++];
  }
  const float* norm_g = (const float*)d_in[ii++];
  const float* norm_b = (const float*)d_in[ii++];
  const float* out_w  = (const float*)d_in[ii++];
  const float* out_b  = (const float*)d_in[ii++];

  // ---- workspace carve ----
  char* p = (char*)d_ws;
  auto alloc = [&](size_t bytes) -> char* {
    char* r = p; p += (bytes + 255) & ~(size_t)255; return r;
  };
  _Float16* XU    = (_Float16*)alloc(4ull * 128 * 128 * 256 * 2);  // upsampled x1, f16
  _Float16* X1UP  = (_Float16*)alloc(65536ull * 128 * 2);          // 1x1-conv out
  _Float16* APE   = XU;                                            // im2col aliases XU
  float*    Pbuf  = (float*)   alloc(4096ull * 192 * 4);
  float*    X     = (float*)   alloc(4096ull * 192 * 4);
  _Float16* XH    = (_Float16*)alloc(4096ull * 192 * 2);
  float*    QKV   = (float*)   alloc(4096ull * 576 * 4);
  _Float16* ATTNH = (_Float16*)alloc(4096ull * 192 * 2);
  _Float16* H1    = (_Float16*)alloc(4096ull * 768 * 2);
  _Float16* UPWp  = (_Float16*)alloc(256ull * 128 * 2);
  _Float16* PEWp  = (_Float16*)alloc(4096ull * 192 * 2);
  _Float16* QKVWp[2], *PROJWp[2], *FC1Wp[2], *FC2Wp[2];
  float* BIAS576[2];
  for (int b = 0; b < 2; ++b) {
    QKVWp[b]  = (_Float16*)alloc(192ull * 576 * 2);
    PROJWp[b] = (_Float16*)alloc(192ull * 192 * 2);
    FC1Wp[b]  = (_Float16*)alloc(192ull * 768 * 2);
    FC2Wp[b]  = (_Float16*)alloc(768ull * 192 * 2);
    BIAS576[b]= (float*)alloc(576 * 4);
  }
  _Float16* OUTWp = (_Float16*)alloc(192ull * 128 * 2);
  float* OUT225   = (float*)alloc(225 * 6 * 4);
  float* RPB      = (float*)alloc(6 * 64 * 64 * 4);

  auto prepack = [&](const float* W, _Float16* Bp, int K, int N) {
    int tot = K * N;
    prepack_b_k<<<(tot + 255) / 256, 256, 0, stream>>>(W, Bp, K, N);
  };
  // nt = N-tiles per wave (A-reuse factor); (N/16) % nt == 0 for all calls.
  auto gemm = [&](const _Float16* A, const _Float16* Bp, const float* bias, void* C,
                  int M, int K, int N, int nt, int act, int outMode) {
    dim3 g(M / 16, (N / 16) / nt);
    switch (nt) {
      case 4:  gemm_wmma_k<4> <<<g, dim3(32), 0, stream>>>(A, Bp, bias, C, M, K, N, act, outMode); break;
      case 6:  gemm_wmma_k<6> <<<g, dim3(32), 0, stream>>>(A, Bp, bias, C, M, K, N, act, outMode); break;
      case 8:  gemm_wmma_k<8> <<<g, dim3(32), 0, stream>>>(A, Bp, bias, C, M, K, N, act, outMode); break;
      default: gemm_wmma_k<12><<<g, dim3(32), 0, stream>>>(A, Bp, bias, C, M, K, N, act, outMode); break;
    }
  };

  // ---- prepack all weights ----
  prepack(up_w, UPWp, 256, 128);
  prepack(pe_w, PEWp, 4096, 192);   // pe_w flat [192][4096] with k=c*16+kh*4+kw
  for (int b = 0; b < 2; ++b) {
    prepack(blk[b].qkv_w, QKVWp[b], 192, 576);
    prepack(blk[b].proj_w, PROJWp[b], 192, 192);
    prepack(blk[b].fc1_w, FC1Wp[b], 192, 768);
    prepack(blk[b].fc2_w, FC2Wp[b], 768, 192);
    pack_bias576_k<<<1, 576, 0, stream>>>(blk[b].q_bias, blk[b].v_bias, BIAS576[b]);
  }
  prepack(out_w, OUTWp, 192, 128);

  // ---- stage 1: upsample + 1x1 conv + im2col + patch-embed + LN ----
  upsample_k<<<65536, 256, 0, stream>>>(x1, XU);
  gemm(XU, UPWp, up_b, X1UP, 65536, 256, 128, 8, 0, 1);     // 4096 blocks
  im2col_k<<<65536, 256, 0, stream>>>(x2, X1UP, APE);
  gemm(APE, PEWp, pe_b, Pbuf, 4096, 4096, 192, 6, 0, 0);    // 512 blocks
  addln_k<<<4096, 64, 0, stream>>>(Pbuf, X, XH, pe_g, pe_bn, 0);

  // ---- stage 2: two SwinV2 blocks ----
  for (int b = 0; b < 2; ++b) {
    cpb_hidden_k<<<225, 512, 0, stream>>>(blk[b].cpb_w1, blk[b].cpb_b1, blk[b].cpb_w2, OUT225);
    rpb_expand_k<<<96, 256, 0, stream>>>(OUT225, RPB);
    gemm(XH, QKVWp[b], BIAS576[b], QKV, 4096, 192, 576, 12, 0, 0);  // 768 blocks
    if (b == 0) attn_k<0><<<384, 64, 0, stream>>>(QKV, RPB, blk[b].logit_scale, ATTNH);
    else        attn_k<4><<<384, 64, 0, stream>>>(QKV, RPB, blk[b].logit_scale, ATTNH);
    gemm(ATTNH, PROJWp[b], blk[b].proj_b, Pbuf, 4096, 192, 192, 6, 0, 0);  // 512 blocks
    addln_k<<<4096, 64, 0, stream>>>(Pbuf, X, XH, blk[b].n1g, blk[b].n1b, 1);
    gemm(XH, FC1Wp[b], blk[b].fc1_b, H1, 4096, 192, 768, 12, 1, 1);  // GELU f16, 1024 blocks
    gemm(H1, FC2Wp[b], blk[b].fc2_b, Pbuf, 4096, 768, 192, 6, 0, 0); // 512 blocks
    addln_k<<<4096, 64, 0, stream>>>(Pbuf, X, XH, blk[b].n2g, blk[b].n2b, 1);
  }

  // ---- stage 3: final LN + output projection (NCHW scatter into d_out) ----
  addln_k<<<4096, 64, 0, stream>>>(X, X, XH, norm_g, norm_b, 0);
  gemm(XH, OUTWp, out_b, d_out, 4096, 192, 128, 4, 0, 2);   // 512 blocks
}
